// MultiStageHead_66958540144651
// MI455X (gfx1250) — matrix operand order repaired and checked by reading, not compiled, not run on hardware
//
#include <hip/hip_runtime.h>
#include <hip/hip_bf16.h>

// -------- CDNA5 WMMA / vector types --------
typedef __attribute__((ext_vector_type(16))) __bf16 v16bf;
typedef __attribute__((ext_vector_type(8)))  __bf16 v8bf;
typedef __attribute__((ext_vector_type(4)))  __bf16 v4bf;
typedef __attribute__((ext_vector_type(8)))  float  v8f;
typedef __attribute__((ext_vector_type(4)))  float  v4f;
typedef __attribute__((ext_vector_type(4)))  int    v4i;

#define KCHUNK 128
#define TB 256            // 8 wave32s per block

// gfx1250 async global->LDS path (guarded: falls back to sync staging if the
// toolchain doesn't declare the builtins)
#if defined(__gfx1250__) &&                                        \
    __has_builtin(__builtin_amdgcn_global_load_async_to_lds_b128) && \
    __has_builtin(__builtin_amdgcn_s_wait_asynccnt)
#define USE_ASYNC_LDS 1
typedef __attribute__((address_space(1))) v4i* gptr_b128;
typedef __attribute__((address_space(3))) v4i* lptr_b128;
#else
#define USE_ASYNC_LDS 0
#endif

// ---------------------------------------------------------------------------
// Split fp32 -> (bf16_hi, bf16_lo) planes.  a = hi + lo + O(2^-18 * a).
// ---------------------------------------------------------------------------
__global__ void __launch_bounds__(TB)
split_kernel(const float* __restrict__ X, __bf16* __restrict__ hi,
             __bf16* __restrict__ lo, int n4)
{
    const int i = blockIdx.x * TB + threadIdx.x;
    if (i >= n4) return;
    const v4f v = ((const v4f*)X)[i];
    v4bf h, l;
#pragma unroll
    for (int e = 0; e < 4; ++e) {
        const __bf16 hh = (__bf16)v[e];
        h[e] = hh;
        l[e] = (__bf16)(v[e] - (float)hh);
    }
    ((v4bf*)hi)[i] = h;
    ((v4bf*)lo)[i] = l;
}

// ---------------------------------------------------------------------------
// Core: one wave computes a 16x16 fp32 tile of A(MxK) @ B(KxN) with the
// bf16x3 split (Ahi*Bhi + Ahi*Blo + Alo*Bhi) on v_wmma_f32_16x16x32_bf16.
// A is pre-split into bf16 hi/lo planes (no conversions in the hot loop).
// B (fp32) is staged into LDS per block:
//   - async path: double-buffered GLOBAL_LOAD_ASYNC_TO_LDS_B128 pipeline
//     (next chunk's HBM fetch overlaps current chunk's WMMAs), then an
//     LDS->LDS hi/lo split transposed to [n][k].
//   - sync path:  global->VGPR->split->LDS (round-2 behavior).
// NT: non-temporal policy for B on the sync path (read-once streaming).
// ---------------------------------------------------------------------------
template <bool NT>
__device__ __forceinline__ v8f gemm_tile_bf16x3(const __bf16* __restrict__ Ahi,
                                                const __bf16* __restrict__ Alo,
                                                const float* __restrict__ B,
                                                int rowA, int n0, int N, int K)
{
    __shared__ __bf16 Bs_hi[16 * KCHUNK] __attribute__((aligned(32)));
    __shared__ __bf16 Bs_lo[16 * KCHUNK] __attribute__((aligned(32)));

    const int t    = threadIdx.x;
    const int lane = t & 31;
    const int half = lane >> 4;   // lanes 0-15 vs 16-31
    const int lm   = lane & 15;

    v8f acc = {};

#if USE_ASYNC_LDS
    __shared__ float Braw[2][KCHUNK * 16] __attribute__((aligned(16)));  // 2 x 8 KB

    // Issue one chunk's raw fp32 panel: 2 x b128 per thread, ASYNCcnt-tracked.
    auto issue_chunk = [&](int kc, int buf) {
#pragma unroll
        for (int i = 0; i < 2; ++i) {
            const int f  = t + TB * i;          // v4f index 0..511
            const int kk = f >> 2;              // k within chunk
            const int c4 = (f & 3) << 2;        // col group
            const float* g = B + (size_t)(kc + kk) * N + n0 + c4;
            float* l = &Braw[buf][kk * 16 + c4];
            __builtin_amdgcn_global_load_async_to_lds_b128(
                (gptr_b128)g, (lptr_b128)l, 0, 0);
        }
    };

    issue_chunk(0, 0);
    int buf = 0;
    for (int kc = 0; kc < K; kc += KCHUNK) {
        const bool more = (kc + KCHUNK < K);   // wave-uniform
        if (more) {
            issue_chunk(kc + KCHUNK, buf ^ 1); // prefetch next panel
            __builtin_amdgcn_s_wait_asynccnt(2);   // retire current panel only
        } else {
            __builtin_amdgcn_s_wait_asynccnt(0);
        }
        __syncthreads();

        // LDS->LDS: split fp32 panel into bf16 hi/lo, transposed [n][k]
#pragma unroll
        for (int i = 0; i < 2; ++i) {
            const int f  = t + TB * i;
            const int kk = f >> 2;
            const int c4 = (f & 3) << 2;
            const v4f v4 = *(const v4f*)&Braw[buf][kk * 16 + c4];
#pragma unroll
            for (int e = 0; e < 4; ++e) {
                const float  f32 = v4[e];
                const __bf16 hi  = (__bf16)f32;
                Bs_hi[(c4 + e) * KCHUNK + kk] = hi;
                Bs_lo[(c4 + e) * KCHUNK + kk] = (__bf16)(f32 - (float)hi);
            }
        }
        __syncthreads();
#else
    for (int kc = 0; kc < K; kc += KCHUNK) {
        // ---- cooperative load of B[kc..kc+127][n0..n0+15], split hi/lo,
        //      stored transposed [n][k] so fragments are contiguous in LDS
#pragma unroll
        for (int i = 0; i < (KCHUNK * 16) / (4 * TB); ++i) {  // 2 x v4f / thread
            const int f  = t + TB * i;          // v4f index 0..511
            const int kk = f >> 2;              // k within chunk 0..127
            const int c4 = (f & 3) << 2;        // col group 0,4,8,12
            const v4f* p4 = (const v4f*)(B + (size_t)(kc + kk) * N + n0 + c4);
            const v4f v4 = NT ? __builtin_nontemporal_load(p4) : *p4;
#pragma unroll
            for (int e = 0; e < 4; ++e) {
                const float  f32 = v4[e];
                const __bf16 hi  = (__bf16)f32;
                Bs_hi[(c4 + e) * KCHUNK + kk] = hi;
                Bs_lo[(c4 + e) * KCHUNK + kk] = (__bf16)(f32 - (float)hi);
            }
        }
        if (kc + KCHUNK < K)  // stream next B chunk toward caches
            __builtin_prefetch(B + (size_t)(kc + KCHUNK + (t & (KCHUNK - 1))) * N + n0, 0, 0);
        __syncthreads();
#endif

        // ---- per-wave WMMA over this K chunk (K=32 per step)
        // A fragment (contiguous k order, same as B): K = 16*half + 0..15
        const __bf16* Ah = Ahi + (size_t)rowA * K + kc + 16 * half;
        const __bf16* Al = Alo + (size_t)rowA * K + kc + 16 * half;
#pragma unroll
        for (int kb = 0; kb < KCHUNK; kb += 32) {
            const v8bf ah0 = *(const v8bf*)(Ah + kb);
            const v8bf ah1 = *(const v8bf*)(Ah + kb + 8);
            const v8bf al0 = *(const v8bf*)(Al + kb);
            const v8bf al1 = *(const v8bf*)(Al + kb + 8);
            const v16bf ah = __builtin_shufflevector(ah0, ah1, 0, 1, 2, 3, 4, 5, 6, 7,
                                                     8, 9, 10, 11, 12, 13, 14, 15);
            const v16bf al = __builtin_shufflevector(al0, al1, 0, 1, 2, 3, 4, 5, 6, 7,
                                                     8, 9, 10, 11, 12, 13, 14, 15);
            // B fragment: K = 16*half + 0..15 at column lm -> contiguous in [n][k] LDS
            const v16bf bh = *(const v16bf*)&Bs_hi[lm * KCHUNK + kb + 16 * half];
            const v16bf bl = *(const v16bf*)&Bs_lo[lm * KCHUNK + kb + 16 * half];

            acc = __builtin_amdgcn_wmma_f32_16x16x32_bf16(false, ah, false, bh, (short)0, acc, false, false);
            acc = __builtin_amdgcn_wmma_f32_16x16x32_bf16(false, ah, false, bl, (short)0, acc, false, false);
            acc = __builtin_amdgcn_wmma_f32_16x16x32_bf16(false, al, false, bh, (short)0, acc, false, false);
        }
#if USE_ASYNC_LDS
        buf ^= 1;            // next iteration's leading wait+barrier fences Bs reuse
#else
        __syncthreads();
#endif
    }
    return acc;
}

// ---------------------------------------------------------------------------
// GEMM + bias.  MODE 0: fp32 out.  MODE 1: relu, then write bf16 hi/lo planes
// (feeds the next GEMM's A operand directly -- h never exists in fp32).
// grid = (N/16, M/128), block = 256.
// ---------------------------------------------------------------------------
template <int MODE, bool NT>
__global__ void __launch_bounds__(TB)
wmma_gemm_kernel(const __bf16* __restrict__ Ahi, const __bf16* __restrict__ Alo,
                 const float* __restrict__ B, const float* __restrict__ bias,
                 float* __restrict__ out, __bf16* __restrict__ outHi,
                 __bf16* __restrict__ outLo, int N, int K)
{
    const int t    = threadIdx.x;
    const int lane = t & 31;
    const int half = lane >> 4;
    const int lm   = lane & 15;
    const int n0      = blockIdx.x * 16;
    const int rowBase = blockIdx.y * 128 + (t >> 5) * 16;

    v8f acc = gemm_tile_bf16x3<NT>(Ahi, Alo, B, rowBase + lm, n0, N, K);

    const float bn = bias[n0 + lm];
#pragma unroll
    for (int j = 0; j < 8; ++j) {           // C layout: VGPR j -> M = j + 8*half
        float v = acc[j] + bn;
        const size_t o = (size_t)(rowBase + j + 8 * half) * N + n0 + lm;
        if (MODE == 0) {
            out[o] = v;
        } else {
            v = v > 0.f ? v : 0.f;
            const __bf16 hh = (__bf16)v;
            outHi[o] = hh;
            outLo[o] = (__bf16)(v - (float)hh);
        }
    }
}

// ---------------------------------------------------------------------------
// Fused video FFN + classifier: hidden tile never hits memory.  Each relu'd
// element hidden[m][n] contributes val * clf_W[n][c] to P[m][n/1024][c].
// In-wave shfl_xor reduction over the 16 columns, then float global atomics.
// grid = (3072/16, 1024/128) = (192, 8), block = 256.
// ---------------------------------------------------------------------------
__global__ void __launch_bounds__(TB)
wmma_ffn_clf_kernel(const __bf16* __restrict__ Ahi, const __bf16* __restrict__ Alo,
                    const float* __restrict__ B, const float* __restrict__ bias,
                    const float* __restrict__ clfW, float* __restrict__ P,
                    int N, int K)
{
    const int t    = threadIdx.x;
    const int lane = t & 31;
    const int half = lane >> 4;
    const int lm   = lane & 15;
    const int n0      = blockIdx.x * 16;
    const int rowBase = blockIdx.y * 128 + (t >> 5) * 16;

    v8f acc = gemm_tile_bf16x3<false>(Ahi, Alo, B, rowBase + lm, n0, N, K);

    const float bn  = bias[n0 + lm];
    const int   n   = n0 + lm;
    const int   seg = n0 >> 10;             // 16-col tile never straddles a 1024 segment
    const float w0 = clfW[n * 3 + 0];
    const float w1 = clfW[n * 3 + 1];
    const float w2 = clfW[n * 3 + 2];

#pragma unroll
    for (int j = 0; j < 8; ++j) {
        float v = acc[j] + bn;
        v = v > 0.f ? v : 0.f;              // relu(hidden)
        float s0 = v * w0, s1 = v * w1, s2 = v * w2;
#pragma unroll
        for (int m = 1; m < 16; m <<= 1) {  // butterfly within each 16-lane half
            s0 += __shfl_xor(s0, m, 32);
            s1 += __shfl_xor(s1, m, 32);
            s2 += __shfl_xor(s2, m, 32);
        }
        if (lm == 0) {
            const int row = rowBase + j + 8 * half;
            atomicAdd(&P[(row * 3 + seg) * 3 + 0], s0);
            atomicAdd(&P[(row * 3 + seg) * 3 + 1], s1);
            atomicAdd(&P[(row * 3 + seg) * 3 + 2], s2);
        }
    }
}

// Zero the P accumulator (1024 rows * 3 segs * 3 ch = 9216 floats).
__global__ void init_p_kernel(float* __restrict__ P)
{
    for (int i = threadIdx.x; i < 1024 * 3 * 3; i += TB) P[i] = 0.f;
}

// ---------------------------------------------------------------------------
// moment_logits[b,i,j,c] = p_st[b,i,c] + p_ed[b,j,c] + p_mid_all[b,(i+j)/2,c] + clf_b[c]
// pred_bds[b,i,j,0] = i + ml0 ; pred_bds[b,i,j,1] = j + ml1
// ---------------------------------------------------------------------------
__global__ void __launch_bounds__(TB)
moment_kernel(const float* __restrict__ P, const float* __restrict__ clfb,
              float* __restrict__ ml, float* __restrict__ pb)
{
    const int idx = blockIdx.x * TB + threadIdx.x;   // 0..262143 over (b,i,j)
    const int b = idx >> 16;
    const int r = idx & 65535;
    const int i = r >> 8;
    const int j = r & 255;
    const int mid = (i + j) >> 1;
    const int mi = b * 256 + i, mj = b * 256 + j, mm = b * 256 + mid;

    const float l0 = P[(mi * 3 + 0) * 3 + 0] + P[(mj * 3 + 1) * 3 + 0] + P[(mm * 3 + 2) * 3 + 0] + clfb[0];
    const float l1 = P[(mi * 3 + 0) * 3 + 1] + P[(mj * 3 + 1) * 3 + 1] + P[(mm * 3 + 2) * 3 + 1] + clfb[1];
    const float l2 = P[(mi * 3 + 0) * 3 + 2] + P[(mj * 3 + 1) * 3 + 2] + P[(mm * 3 + 2) * 3 + 2] + clfb[2];

    const size_t o3 = (size_t)idx * 3;
    ml[o3 + 0] = l0; ml[o3 + 1] = l1; ml[o3 + 2] = l2;
    const size_t o2 = (size_t)idx * 2;
    pb[o2 + 0] = (float)i + l0;
    pb[o2 + 1] = (float)j + l1;
}

// ---------------------------------------------------------------------------
extern "C" void kernel_launch(void* const* d_in, const int* in_sizes, int n_in,
                              void* d_out, int out_size, void* d_ws, size_t ws_size,
                              hipStream_t stream)
{
    const float* vid  = (const float*)d_in[0];  // (4,256,1024)
    const float* txt  = (const float*)d_in[1];  // (4,32,1024)
    const float* W1   = (const float*)d_in[2];  // (1024,1024)
    const float* b1   = (const float*)d_in[3];  // (1024,)
    const float* W2   = (const float*)d_in[4];  // (1024,32000)
    const float* b2   = (const float*)d_in[5];  // (32000,)
    const float* ffnW = (const float*)d_in[6];  // (1024,3072)
    const float* ffnb = (const float*)d_in[7];  // (3072,)
    const float* clfW = (const float*)d_in[8];  // (3072,3)
    const float* clfb = (const float*)d_in[9];  // (3,)

    float* out  = (float*)d_out;
    float* word = out;                    // 4*32*32000   = 4,096,000
    float* ml   = out + 4096000;          // 4*256*256*3  =   786,432
    float* pb   = out + 4882432;          // 4*256*256*2  =   524,288

    // workspace: bf16 hi/lo planes for all A operands + P accumulator (~5.3 MB)
    const int TXT_N = 128 * 1024;         // txt rows x K
    const int VID_N = 1024 * 1024;        // vid rows x K
    __bf16* txtHi = (__bf16*)d_ws;
    __bf16* txtLo = txtHi + TXT_N;
    __bf16* vidHi = txtLo + TXT_N;
    __bf16* vidLo = vidHi + VID_N;
    __bf16* hHi   = vidLo + VID_N;
    __bf16* hLo   = hHi + TXT_N;
    float*  P     = (float*)(hLo + TXT_N);

    // pre-split A operands to bf16 hi/lo (done once; reused by every n-tile block)
    split_kernel<<<(TXT_N / 4 + TB - 1) / TB, TB, 0, stream>>>(txt, txtHi, txtLo, TXT_N / 4);
    split_kernel<<<(VID_N / 4 + TB - 1) / TB, TB, 0, stream>>>(vid, vidHi, vidLo, VID_N / 4);

    // text: h = relu(txt@W1+b1), written directly as bf16 hi/lo planes
    wmma_gemm_kernel<1, false><<<dim3(1024 / 16, 1), TB, 0, stream>>>(
        txtHi, txtLo, W1, b1, nullptr, hHi, hLo, 1024, 1024);
    // text: word_logits = h@W2+b2  (dominant: 131 MB of W2, async-pipelined)
    wmma_gemm_kernel<0, true><<<dim3(32000 / 16, 1), TB, 0, stream>>>(
        hHi, hLo, W2, b2, word, nullptr, nullptr, 32000, 1024);

    // video: fused relu(vid@ffnW+b) -> segmented clf projections into P
    init_p_kernel<<<1, TB, 0, stream>>>(P);
    wmma_ffn_clf_kernel<<<dim3(3072 / 16, 1024 / 128), TB, 0, stream>>>(
        vidHi, vidLo, ffnW, ffnb, clfW, P, 3072, 1024);

    // assemble moment_logits + pred_bds
    moment_kernel<<<(4 * 256 * 256) / TB, TB, 0, stream>>>(P, clfb, ml, pb);
}
